// gFFNGlobal_21225728376931
// MI455X (gfx1250) — matrix operand order repaired and checked by reading, not compiled, not run on hardware
//
#include <hip/hip_runtime.h>

// ---------------- problem constants ----------------
#define GB 2
#define GL 2048
#define GD 2048
#define GM (GB * GL)   // 4096 rows of the GEMM
#define GN 2048        // dim_out
#define GK 18432       // cat_dim = 9*GD

// ---------------- GEMM tiling ----------------
#define BM 128
#define BN 128
#define BK 64
#define LDT 72         // padded LDS row stride in elements (128B data + 16B pad)

typedef __attribute__((ext_vector_type(16))) __bf16 v16bf;
typedef __attribute__((ext_vector_type(8)))  float  v8f;
typedef unsigned int u32x4 __attribute__((ext_vector_type(4)));
typedef int          i32x8 __attribute__((ext_vector_type(8)));
typedef int          i32x4 __attribute__((ext_vector_type(4)));

#if __has_builtin(__builtin_amdgcn_tensor_load_to_lds)
#define USE_TDM 1
#else
#define USE_TDM 0
#endif

union Frag16 { uint4 q[2]; v16bf v; };

static __device__ __forceinline__ unsigned short f32_to_bf16(float f) {
    union { float f; unsigned int u; } v; v.f = f;
    unsigned int u = v.u;
    return (unsigned short)((u + 0x7FFFu + ((u >> 16) & 1u)) >> 16); // RNE
}

static __device__ __forceinline__ float silu_f(float x) {
    return x / (1.0f + __expf(-x));
}

static __device__ __forceinline__ void wait_tensorcnt0() {
#if __has_builtin(__builtin_amdgcn_s_wait_tensorcnt)
    __builtin_amdgcn_s_wait_tensorcnt(0);
#else
    asm volatile("s_wait_tensorcnt 0x0" ::: "memory");
#endif
}

#if USE_TDM
// Issue one TDM descriptor: load a BM x BK bf16 tile (row stride GK elements)
// from global into LDS, hardware-padding each 128B row with 16B (-> LDT=72).
static __device__ __forceinline__ void tdm_tile_load(const unsigned short* gsrc,
                                                     const unsigned short* ldst,
                                                     unsigned tensorRows) {
    const unsigned long long ga = (unsigned long long)(uintptr_t)gsrc;
    const unsigned lds = (unsigned)(uintptr_t)ldst;   // low 32 bits = LDS offset
    u32x4 g0;
    g0[0] = 1u;                                        // count=1 (valid descriptor)
    g0[1] = lds;                                       // lds_addr
    g0[2] = (unsigned)ga;                              // global_addr[31:0]
    g0[3] = (unsigned)((ga >> 32) & 0x01FFFFFFull)     // global_addr[56:32]
          | (2u << 30);                                // type=2 ("image")
    i32x8 g1;
    g1[0] = (int)((1u << 16)                           // data_size = 2 bytes
                | (1u << 20)                           // pad_enable
                | (4u << 22)                           // pad_interval: 32 dwords (128B)
                | (3u << 25));                         // pad_amount: 4 dwords (16B)
    g1[1] = (int)(((unsigned)GK & 0xFFFFu) << 16);     // tensor_dim0[15:0]
    g1[2] = (int)((((unsigned)GK >> 16) & 0xFFFFu)     // tensor_dim0[31:16]
                | ((tensorRows & 0xFFFFu) << 16));     // tensor_dim1[15:0]
    g1[3] = (int)(((tensorRows >> 16) & 0xFFFFu)       // tensor_dim1[31:16]
                | ((unsigned)BK << 16));               // tile_dim0 = 64
    g1[4] = (int)(unsigned)BM;                         // tile_dim1 = 128 (tile_dim2=0)
    g1[5] = (int)(unsigned)GK;                         // tensor_dim0_stride[31:0]
    g1[6] = 0;                                         // stride hi / dim1_stride lo
    g1[7] = 0;
    const i32x4 z4 = {0, 0, 0, 0};                     // 2D tile: groups 2/3 unused
    const i32x8 z8 = {0, 0, 0, 0, 0, 0, 0, 0};
    __builtin_amdgcn_tensor_load_to_lds(g0, g1, z4, z4, z8, 0);
}
#endif

// ---------------- kernel 1: glo = mean over L ----------------
__global__ void glo_kernel(const float* __restrict__ x, float* __restrict__ glo) {
    const int idx = blockIdx.x * blockDim.x + threadIdx.x;   // 0 .. GB*GD-1
    const int b = idx / GD, d = idx % GD;
    const float* p = x + (size_t)b * GL * GD + d;
    float s = 0.0f;
    for (int l = 0; l < GL; ++l) s += p[(size_t)l * GD];
    glo[idx] = s * (1.0f / (float)GL);
}

// ---------------- kernel 2: build mixed features in bf16 ----------------
__global__ void mixed_kernel(const float* __restrict__ x, const float* __restrict__ glo,
                             unsigned short* __restrict__ mixed) {
    const int HALF = GK / 2;                                  // 9216 pairs per row
    const size_t p = (size_t)blockIdx.x * blockDim.x + threadIdx.x;
    const int m = (int)(p / HALF);
    const int c = (int)(p % HALF) * 2;                        // even column
    const int f = c / GD;                                     // feature block 0..8
    const int d = c % GD;
    const int b = m / GL;
    const float* xr = x + (size_t)m * GD;
    const float* gr = glo + (size_t)b * GD;
    const float x0 = xr[d], x1 = xr[d + 1];
    float y0, y1;
    if (f == 0) {
        y0 = silu_f(x0); y1 = silu_f(x1);
    } else {
        const int shifts[4] = {1, 2, 4, 8};
        const int s = shifts[(f - 1) >> 1];
        int d0 = d - s;     if (d0 < 0) d0 += GD;
        int d1 = d + 1 - s; if (d1 < 0) d1 += GD;
        const float g0 = gr[d0], g1 = gr[d1];
        if (f & 1) {                                          // inner(s)
            y0 = silu_f(x0 * g0); y1 = silu_f(x1 * g1);
        } else {                                              // wedge(s)
            y0 = x0 * g0 - xr[d0] * gr[d];
            y1 = x1 * g1 - xr[d1] * gr[d + 1];
        }
    }
    const unsigned int packed =
        (unsigned int)f32_to_bf16(y0) | ((unsigned int)f32_to_bf16(y1) << 16);
    *(unsigned int*)(mixed + (size_t)m * GK + c) = packed;
}

// ---------------- kernel 3: W fp32 -> bf16 ----------------
__global__ void wconv_kernel(const float* __restrict__ W, unsigned short* __restrict__ Wb) {
    const size_t p = (size_t)blockIdx.x * blockDim.x + threadIdx.x;  // pair index
    const float2 w = *(const float2*)(W + p * 2);
    ((unsigned int*)Wb)[p] =
        (unsigned int)f32_to_bf16(w.x) | ((unsigned int)f32_to_bf16(w.y) << 16);
}

// ---------------- kernel 4: bf16 WMMA GEMM (TDM double-buffered) ----------------
__global__ __launch_bounds__(256)
void gemm_bf16(const unsigned short* __restrict__ Abuf,   // [GM][GK] bf16
               const unsigned short* __restrict__ Bbuf,   // [GN][GK] bf16 (W row-major)
               const float* __restrict__ bias,
               float* __restrict__ out) {                 // [GM][GN]
    __shared__ unsigned short As[2][BM * LDT];
    __shared__ unsigned short Bs[2][BN * LDT];

    const int tid   = threadIdx.x;
    const int lane  = tid & 31;
    const int wid   = tid >> 5;       // 8 waves
    const int waveM = wid >> 2;       // 0..1  (64 rows each)
    const int waveN = wid & 3;        // 0..3  (32 cols each)
    const int mBase = blockIdx.y * BM;
    const int nBase = blockIdx.x * BN;
    const int r  = lane & 15;         // row/col within fragment
    const int hi = lane >> 4;         // lane-half selects K sub-range

    v8f acc[4][2] = {};

    const unsigned short* Arow = Abuf + (size_t)mBase * GK;
    const unsigned short* Brow = Bbuf + (size_t)nBase * GK;

#if USE_TDM
    // ---- prologue: DMA stage 0 ----
    if (wid == 0) {
        tdm_tile_load(Arow, &As[0][0], (unsigned)GM);
        tdm_tile_load(Brow, &Bs[0][0], (unsigned)GN);
        wait_tensorcnt0();
    }
    __syncthreads();
#else
    const int lrow = tid >> 3;        // 0..31
    const int lcol = (tid & 7) * 8;   // 16B granules within BK row
#endif

    int stage = 0;
    for (int kt = 0; kt < GK; kt += BK) {
#if USE_TDM
        if (wid == 0 && kt + BK < GK) {   // DMA next tile into the other stage
            tdm_tile_load(Arow + kt + BK, &As[stage ^ 1][0], (unsigned)GM);
            tdm_tile_load(Brow + kt + BK, &Bs[stage ^ 1][0], (unsigned)GN);
        }
#else
#pragma unroll
        for (int c = 0; c < 4; ++c) {
            const uint4 av = *(const uint4*)(Arow + (size_t)(lrow + c * 32) * GK + kt + lcol);
            *(uint4*)(&As[0][(lrow + c * 32) * LDT + lcol]) = av;
            const uint4 bv = *(const uint4*)(Brow + (size_t)(lrow + c * 32) * GK + kt + lcol);
            *(uint4*)(&Bs[0][(lrow + c * 32) * LDT + lcol]) = bv;
        }
        __syncthreads();
#endif

        const unsigned short* Atile = &As[stage][0];
        const unsigned short* Btile = &Bs[stage][0];
#pragma unroll
        for (int ks = 0; ks < BK; ks += 32) {
            // A fragments: 16x32, lane holds M=r; K chunks at hi*8 and 16+hi*8
            v16bf afrag[4];
#pragma unroll
            for (int fm = 0; fm < 4; ++fm) {
                const unsigned short* p =
                    &Atile[(waveM * 64 + fm * 16 + r) * LDT + ks + hi * 8];
                Frag16 t;
                t.q[0] = *(const uint4*)p;
                t.q[1] = *(const uint4*)(p + 16);
                afrag[fm] = t.v;
            }
            // B fragments: 32x16, lane holds N=r; K chunks at hi*8 and 16+hi*8
            v16bf bfrag[2];
#pragma unroll
            for (int fn = 0; fn < 2; ++fn) {
                const unsigned short* p =
                    &Btile[(waveN * 32 + fn * 16 + r) * LDT + ks + hi * 8];
                Frag16 t;
                t.q[0] = *(const uint4*)p;
                t.q[1] = *(const uint4*)(p + 16);
                bfrag[fn] = t.v;
            }
#pragma unroll
            for (int fm = 0; fm < 4; ++fm)
#pragma unroll
                for (int fn = 0; fn < 2; ++fn)
                    acc[fm][fn] = __builtin_amdgcn_wmma_f32_16x16x32_bf16(
                        false, afrag[fm], false, bfrag[fn],
                        (short)0, acc[fm][fn], false, false);
        }

#if USE_TDM
        if (wid == 0) wait_tensorcnt0();   // next stage fully in LDS
        __syncthreads();
        stage ^= 1;
#else
        __syncthreads();
#endif
    }

    // epilogue: C/D layout -> VGPR i holds row hi*8+i, lane r holds column
#pragma unroll
    for (int fm = 0; fm < 4; ++fm) {
#pragma unroll
        for (int fn = 0; fn < 2; ++fn) {
            const int m0 = mBase + waveM * 64 + fm * 16 + hi * 8;
            const int n  = nBase + waveN * 32 + fn * 16 + r;
            const float bv = bias[n];
            float* op = out + (size_t)m0 * GN + n;
#pragma unroll
            for (int i = 0; i < 8; ++i)
                op[(size_t)i * GN] = acc[fm][fn][i] + bv;
        }
    }
}

// ---------------- host side ----------------
extern "C" void kernel_launch(void* const* d_in, const int* in_sizes, int n_in,
                              void* d_out, int out_size, void* d_ws, size_t ws_size,
                              hipStream_t stream) {
    const float* x    = (const float*)d_in[0];   // [GB, GL, GD]
    const float* W    = (const float*)d_in[1];   // [GN, GK] fp32
    const float* bias = (const float*)d_in[2];   // [GN]
    float* out = (float*)d_out;                  // [GM, GN]

    // workspace layout: glo (16KB, padded to 32KB) | mixed bf16 (151MB) | W bf16 (75.5MB)
    char* ws = (char*)d_ws;
    float* glo = (float*)ws;
    unsigned short* mixed = (unsigned short*)(ws + 32 * 1024);
    unsigned short* Wb    = (unsigned short*)(ws + 32 * 1024 + (size_t)GM * GK * 2);

    glo_kernel<<<(GB * GD) / 256, 256, 0, stream>>>(x, glo);

    const int mixedBlocks = (int)(((size_t)GM * GK / 2) / 256);   // 147456
    mixed_kernel<<<mixedBlocks, 256, 0, stream>>>(x, glo, mixed);

    const int wBlocks = (int)(((size_t)GN * GK / 2) / 256);       // 73728
    wconv_kernel<<<wBlocks, 256, 0, stream>>>(W, Wb);

    dim3 grid(GN / BN, GM / BM);                                   // 16 x 32
    gemm_bf16<<<grid, 256, 0, stream>>>(mixed, Wb, bias, out);
}
